// GhostNet_67894843015668
// MI455X (gfx1250) — matrix operand order repaired and checked by reading, not compiled, not run on hardware
//
#include <hip/hip_runtime.h>

typedef __attribute__((ext_vector_type(16))) _Float16 v16h;
typedef __attribute__((ext_vector_type(8)))  _Float16 v8h;
typedef __attribute__((ext_vector_type(8)))  float    v8f;

#define BN_EPS 1e-5f
#define BATCH 32

// ---------------------------------------------------------------------------
// Reduction: per (b,c) mean over HW.  x is [B, C, HW] with batch stride bS.
// ---------------------------------------------------------------------------
__global__ void k_mean_hw(const float* __restrict__ x, long bS, int C, int HW,
                          float* __restrict__ out) {
  int bc = blockIdx.x;
  int b = bc / C, c = bc % C;
  const float* p = x + (long)b * bS + (long)c * HW;
  float s = 0.f;
  for (int i = threadIdx.x; i < HW; i += blockDim.x) s += p[i];
  __shared__ float red[256];
  red[threadIdx.x] = s;
  __syncthreads();
  for (int off = 128; off > 0; off >>= 1) {
    if ((int)threadIdx.x < off) red[threadIdx.x] += red[threadIdx.x + off];
    __syncthreads();
  }
  if (threadIdx.x == 0) out[bc] = red[0] / (float)HW;
}

// ---------------------------------------------------------------------------
// Small FC: out[b,o] = act( in[b,:] . W[o,:] + bias[o] )
// act: 0 none, 1 relu, 2 sigmoid (router), 3 hard_sigmoid (SE gate)
// ---------------------------------------------------------------------------
__global__ void k_fc(const float* __restrict__ in, const float* __restrict__ W,
                     const float* __restrict__ bias, float* __restrict__ out,
                     int B, int I, int O, int act) {
  int idx = blockIdx.x * blockDim.x + threadIdx.x;
  if (idx >= B * O) return;
  int b = idx / O, o = idx % O;
  float s = bias ? bias[o] : 0.f;
  const float* ip = in + (long)b * I;
  const float* wp = W + (long)o * I;
  for (int i = 0; i < I; ++i) s += ip[i] * wp[i];
  if (act == 1)      s = fmaxf(s, 0.f);
  else if (act == 2) s = 1.f / (1.f + expf(-s));
  else if (act == 3) s = fminf(fmaxf(s + 3.f, 0.f), 6.f) * (1.f / 6.f);
  out[idx] = s;
}

// ---------------------------------------------------------------------------
// Expert blend: out[b,j] = sum_e r[b,e] * W[e,j]   (f16 and f32 variants)
// ---------------------------------------------------------------------------
__global__ void k_blend_h(const float* __restrict__ r, const float* __restrict__ W,
                          _Float16* __restrict__ out, int B, long N) {
  long idx = (long)blockIdx.x * blockDim.x + threadIdx.x;
  if (idx >= (long)B * N) return;
  int b = (int)(idx / N);
  long j = idx % N;
  const float* rp = r + b * 4;
  float s = rp[0] * W[j] + rp[1] * W[N + j] + rp[2] * W[2 * N + j] + rp[3] * W[3 * N + j];
  out[idx] = (_Float16)s;
}

__global__ void k_blend_f(const float* __restrict__ r, const float* __restrict__ W,
                          float* __restrict__ out, int B, long N) {
  long idx = (long)blockIdx.x * blockDim.x + threadIdx.x;
  if (idx >= (long)B * N) return;
  int b = (int)(idx / N);
  long j = idx % N;
  const float* rp = r + b * 4;
  out[idx] = rp[0] * W[j] + rp[1] * W[N + j] + rp[2] * W[2 * N + j] + rp[3] * W[3 * N + j];
}

// ---------------------------------------------------------------------------
// Per-sample 1x1 conv as batched GEMM via WMMA f32_16x16x32_f16.
//   A = blended weights wb[b]  : Cout x Cin  (f16, per sample)
//   B = activations   x[b]     : Cin  x HW   (f32 -> f16 on LDS stage)
//   D = y[b]                   : Cout x HW   (fused BN (+ReLU))
// 128-thread block = 4 waves; each wave owns TWO 16x16 N-tiles (32 HW cols),
// 2 WMMAs per K-step reusing one A fragment.  B tile stored n-major in LDS so
// fragments are contiguous 8-half chunks -> ds_load_b128; edge paths use
// clamped addresses + select (no exec divergence).
// ---------------------------------------------------------------------------
__global__ void k_dc1x1_wmma(const float* __restrict__ x, long inB, int Cin, int HW,
                             const _Float16* __restrict__ wb,
                             float* __restrict__ y, long outB, int Cout,
                             const float* __restrict__ g, const float* __restrict__ bt,
                             const float* __restrict__ mn, const float* __restrict__ vr,
                             int relu) {
  __shared__ _Float16 As[16 * 32] __attribute__((aligned(32)));      // [o][k]
  __shared__ _Float16 Bs[4][32 * 32] __attribute__((aligned(32)));   // [n][k] (transposed)

  int wave = threadIdx.x >> 5;
  int lane = threadIdx.x & 31;
  int b    = blockIdx.z;
  int o0   = blockIdx.y * 16;
  int n0   = (blockIdx.x * 4 + wave) * 32;

  const _Float16* wbp = wb + (long)b * Cout * Cin;
  const float*    xp  = x + (long)b * inB;

  int  mrow  = lane & 15;        // M (A) / N (B,C) index within half-wave
  int  hi    = lane >> 4;        // half-wave selector
  bool aFull = (o0 + 16 <= Cout);

  v8f acc0 = {}, acc1 = {};
  for (int k0 = 0; k0 < Cin; k0 += 32) {
    bool kFull = (k0 + 32 <= Cin);

    // ---- Stage A tile (16 x 32): lanes contiguous along K (coalesced) ----
    if (aFull && kFull) {
#pragma unroll
      for (int rr = 0; rr < 4; ++rr) {
        int r = wave + rr * 4;
        As[r * 32 + lane] = wbp[(long)(o0 + r) * Cin + (k0 + lane)];
      }
    } else {
#pragma unroll
      for (int rr = 0; rr < 4; ++rr) {
        int r = wave + rr * 4;
        int o = o0 + r, k = k0 + lane;
        int oc = o < Cout ? o : Cout - 1;
        int kc = k < Cin ? k : Cin - 1;
        _Float16 val = wbp[(long)oc * Cin + kc];   // clamped: always valid
        As[r * 32 + lane] = (o < Cout && k < Cin) ? val : (_Float16)0.f;
      }
    }

    // ---- Stage per-wave B tile transposed: lane owns column n=n0+lane ----
    _Float16 tmp[32];
    if (kFull && (n0 + 32 <= HW)) {
#pragma unroll
      for (int kk = 0; kk < 32; ++kk)
        tmp[kk] = (_Float16)xp[(long)(k0 + kk) * HW + n0 + lane];
    } else {
      int n = n0 + lane;
      int ncl = n < HW ? n : HW - 1;
#pragma unroll
      for (int kk = 0; kk < 32; ++kk) {
        int kg = k0 + kk;
        int kcl = kg < Cin ? kg : Cin - 1;
        float vv = xp[(long)kcl * HW + ncl];       // clamped: always valid
        tmp[kk] = (kg < Cin && n < HW) ? (_Float16)vv : (_Float16)0.f;
      }
    }
    {
      v8h* brow = (v8h*)&Bs[wave][lane * 32];      // lane-private 64B row
#pragma unroll
      for (int j = 0; j < 4; ++j) {
        v8h ch;
#pragma unroll
        for (int e = 0; e < 8; ++e) ch[e] = tmp[j * 8 + e];
        brow[j] = ch;
      }
    }
    if (kFull && k0 + 32 < Cin)
      __builtin_prefetch(xp + (long)(k0 + 32 + lane) * HW + n0, 0, 1);
    __syncthreads();

    // ---- Fragments: contiguous 8-half chunks -> wide LDS loads ----
    const v8h* Av = (const v8h*)As;
    v8h a_lo = Av[mrow * 4 + hi];                  // k = hi*8 .. hi*8+7
    v8h a_hi = Av[mrow * 4 + 2 + hi];              // k = 16+hi*8 ..
    v16h a = __builtin_shufflevector(a_lo, a_hi,
        0, 1, 2, 3, 4, 5, 6, 7, 8, 9, 10, 11, 12, 13, 14, 15);

    const v8h* Bv = (const v8h*)Bs[wave];
    v8h b0l = Bv[mrow * 4 + hi];
    v8h b0h = Bv[mrow * 4 + 2 + hi];
    v8h b1l = Bv[(mrow + 16) * 4 + hi];
    v8h b1h = Bv[(mrow + 16) * 4 + 2 + hi];
    v16h b0 = __builtin_shufflevector(b0l, b0h,
        0, 1, 2, 3, 4, 5, 6, 7, 8, 9, 10, 11, 12, 13, 14, 15);
    v16h b1 = __builtin_shufflevector(b1l, b1h,
        0, 1, 2, 3, 4, 5, 6, 7, 8, 9, 10, 11, 12, 13, 14, 15);

    acc0 = __builtin_amdgcn_wmma_f32_16x16x32_f16(
        false, a, false, b0, (short)0, acc0, false, false);
    acc1 = __builtin_amdgcn_wmma_f32_16x16x32_f16(
        false, a, false, b1, (short)0, acc1, false, false);
    __syncthreads();
  }

  // ---- Epilogue: BN (+ReLU); C/D layout lane<16: M=r, lane>=16: M=r+8 ----
  float* yp = y + (long)b * outB;
#pragma unroll
  for (int r = 0; r < 8; ++r) {
    int o = o0 + r + (hi ? 8 : 0);
    if (o >= Cout) continue;
    float sc  = g[o] * rsqrtf(vr[o] + BN_EPS);
    float off = bt[o] - mn[o] * sc;
    int n1 = n0 + mrow, n2 = n0 + 16 + mrow;
    float v1 = acc0[r] * sc + off;
    float v2 = acc1[r] * sc + off;
    if (relu) { v1 = fmaxf(v1, 0.f); v2 = fmaxf(v2, 0.f); }
    if (n1 < HW) yp[(long)o * HW + n1] = v1;
    if (n2 < HW) yp[(long)o * HW + n2] = v2;
  }
}

// ---------------------------------------------------------------------------
// Depthwise KxK conv (+BN, +optional ReLU). Weights either shared [C,K,K] or
// per-sample [B,Cw,K,K]. Computes first CoutUsed channels only (ghost cheap op).
// ---------------------------------------------------------------------------
__global__ void k_dwconv(const float* __restrict__ x, long inB, int Cw, int H, int W,
                         const float* __restrict__ wS, const float* __restrict__ wP,
                         int K, int stride, int pad,
                         float* __restrict__ y, long outB, int Ho, int Wo, int CoutUsed,
                         const float* __restrict__ g, const float* __restrict__ bt,
                         const float* __restrict__ mn, const float* __restrict__ vr,
                         int relu, long total) {
  long idx = (long)blockIdx.x * blockDim.x + threadIdx.x;
  if (idx >= total) return;
  int hw = Ho * Wo;
  int per = CoutUsed * hw;
  int b = (int)(idx / per);
  int rem = (int)(idx % per);
  int c = rem / hw, p2 = rem % hw;
  int oh = p2 / Wo, ow = p2 % Wo;

  const float* wptr = wP ? (wP + ((long)b * Cw + c) * K * K) : (wS + (long)c * K * K);
  const float* xc = x + (long)b * inB + (long)c * H * W;
  int ih0 = oh * stride - pad, iw0 = ow * stride - pad;
  float s = 0.f;
  for (int kh = 0; kh < K; ++kh) {
    int ih = ih0 + kh;
    if (ih < 0 || ih >= H) continue;
    for (int kw = 0; kw < K; ++kw) {
      int iw = iw0 + kw;
      if (iw < 0 || iw >= W) continue;
      s += xc[ih * W + iw] * wptr[kh * K + kw];
    }
  }
  float sc = g[c] * rsqrtf(vr[c] + BN_EPS);
  float val = s * sc + (bt[c] - mn[c] * sc);
  if (relu) val = fmaxf(val, 0.f);
  y[(long)b * outB + (long)c * hw + p2] = val;
}

// ---------------------------------------------------------------------------
// Dense dynamic KxK conv (stem: Cin=3). Per-sample weights [B,O,Cin,K,K].
// ---------------------------------------------------------------------------
__global__ void k_convfull(const float* __restrict__ x, int Cin, int H, int W,
                           const float* __restrict__ wP, int K, int stride, int pad,
                           float* __restrict__ y, int O, int Ho, int Wo,
                           const float* __restrict__ g, const float* __restrict__ bt,
                           const float* __restrict__ mn, const float* __restrict__ vr,
                           int relu, long total) {
  long idx = (long)blockIdx.x * blockDim.x + threadIdx.x;
  if (idx >= total) return;
  int hw = Ho * Wo, per = O * hw;
  int b = (int)(idx / per);
  int rem = (int)(idx % per);
  int o = rem / hw, p2 = rem % hw;
  int oh = p2 / Wo, ow = p2 % Wo;
  const float* wp = wP + ((long)b * O + o) * Cin * K * K;
  const float* xb = x + (long)b * Cin * H * W;
  int ih0 = oh * stride - pad, iw0 = ow * stride - pad;
  float s = 0.f;
  for (int ci = 0; ci < Cin; ++ci) {
    const float* xc = xb + (long)ci * H * W;
    const float* wc = wp + ci * K * K;
    for (int kh = 0; kh < K; ++kh) {
      int ih = ih0 + kh;
      if (ih < 0 || ih >= H) continue;
      for (int kw = 0; kw < K; ++kw) {
        int iw = iw0 + kw;
        if (iw < 0 || iw >= W) continue;
        s += xc[ih * W + iw] * wc[kh * K + kw];
      }
    }
  }
  float sc = g[o] * rsqrtf(vr[o] + BN_EPS);
  float val = s * sc + (bt[o] - mn[o] * sc);
  if (relu) val = fmaxf(val, 0.f);
  y[(long)b * per + rem] = val;
}

__global__ void k_scale(float* __restrict__ x, const float* __restrict__ gate,
                        int C, int HW, long total) {
  long idx = (long)blockIdx.x * blockDim.x + threadIdx.x;
  if (idx >= total) return;
  int b = (int)(idx / ((long)C * HW));
  int c = (int)((idx / HW) % C);
  x[idx] *= gate[b * C + c];
}

__global__ void k_add(float* __restrict__ out, const float* __restrict__ a,
                      const float* __restrict__ b, long n) {
  long i = (long)blockIdx.x * blockDim.x + threadIdx.x;
  if (i < n) out[i] = a[i] + b[i];
}

// ---------------------------------------------------------------------------
// Host orchestration
// ---------------------------------------------------------------------------
struct DCp { const float *rw, *rb, *w; };
struct BNp { const float *g, *b, *m, *v; };
struct Ghostp { DCp p; BNp pbn; DCp c; BNp cbn; };
struct Blkp {
  Ghostp g1, g2;
  const float* dw; BNp dwbn;
  const float *sw1, *sb1, *sw2, *sb2;
  DCp s1; BNp s1bn; DCp s2; BNp s2bn;
};
struct BlkCfg { int k, mid, out, rd, s, cin; };
static const BlkCfg BLK[6] = {
  {3, 16, 16, 0,   1, 16},
  {3, 48, 24, 0,   2, 16},
  {5, 72, 40, 20,  2, 24},
  {3, 240, 80, 0,  2, 40},
  {3, 184, 80, 0,  1, 80},
  {5, 672, 160, 168, 2, 80},
};

struct Cur {
  void* const* p; int i;
  const float* n() { return (const float*)p[i++]; }
};
static DCp rdDC(Cur& c) { DCp d; d.rw = c.n(); d.rb = c.n(); d.w = c.n(); return d; }
static BNp rdBN(Cur& c) { BNp d; d.g = c.n(); d.b = c.n(); d.m = c.n(); d.v = c.n(); return d; }
static Ghostp rdG(Cur& c) {
  Ghostp g; g.p = rdDC(c); g.pbn = rdBN(c); g.c = rdDC(c); g.cbn = rdBN(c); return g;
}
static inline dim3 g1d(long n, int t) { return dim3((unsigned int)((n + t - 1) / t)); }

extern "C" void kernel_launch(void* const* d_in, const int* in_sizes, int n_in,
                              void* d_out, int out_size, void* d_ws, size_t ws_size,
                              hipStream_t stream) {
  (void)in_sizes; (void)n_in; (void)out_size; (void)ws_size;
  const int B = BATCH;

  Cur cur{d_in, 0};
  const float* x_in = cur.n();                  // [32,3,224,224]
  DCp stem = rdDC(cur); BNp stembn = rdBN(cur);
  Blkp bp[6];
  for (int i = 0; i < 6; ++i) {
    bp[i].g1 = rdG(cur); bp[i].g2 = rdG(cur);
    if (BLK[i].s > 1) { bp[i].dw = cur.n(); bp[i].dwbn = rdBN(cur); }
    if (BLK[i].rd > 0) { bp[i].sw1 = cur.n(); bp[i].sb1 = cur.n();
                         bp[i].sw2 = cur.n(); bp[i].sb2 = cur.n(); }
    bool sc = !(BLK[i].cin == BLK[i].out && BLK[i].s == 1);
    if (sc) { bp[i].s1 = rdDC(cur); bp[i].s1bn = rdBN(cur);
              bp[i].s2 = rdDC(cur); bp[i].s2bn = rdBN(cur); }
  }
  DCp fin = rdDC(cur); BNp finbn = rdBN(cur);
  const float* head_w = cur.n(); const float* head_b = cur.n();
  const float* cls_w  = cur.n(); const float* cls_b  = cur.n();

  // workspace bump allocator
  char* w8 = (char*)d_ws;
  auto alloc = [&](size_t bytes) -> void* {
    void* p = (void*)w8;
    w8 += (bytes + 255) & ~(size_t)255;
    return p;
  };
  float* ACT0 = (float*)alloc((size_t)6500000 * 4);
  float* ACT1 = (float*)alloc((size_t)6500000 * 4);
  float* MID0 = (float*)alloc((size_t)19500000 * 4);   // ghost1 out at input res
  float* MID1 = (float*)alloc((size_t)5000000 * 4);    // after stride-2 dw
  float* SH0  = (float*)alloc((size_t)3000000 * 4);
  float* SH1  = (float*)alloc((size_t)3000000 * 4);
  float* MEAN = (float*)alloc((size_t)32768 * 4);
  float* ROUT = (float*)alloc((size_t)1024 * 4);
  float* SE1  = (float*)alloc((size_t)8192 * 4);
  float* GATE = (float*)alloc((size_t)32768 * 4);
  _Float16* WBH = (_Float16*)alloc((size_t)4000000 * 2);   // blended 1x1 weights, f16
  float*    WBF = (float*)alloc((size_t)1500000 * 4);      // blended dw/stem weights
  float* POOL = (float*)alloc((size_t)32768 * 4);
  float* FEAT = (float*)alloc((size_t)32 * 1280 * 4);

  // --- dynamic 1x1 conv (routing -> blend f16 -> WMMA GEMM + BN/ReLU) ---
  auto dc1x1 = [&](const float* in, long inB, int Cin, int HW,
                   const DCp& dc, const BNp& bn, int Cout,
                   float* out, long outB, int relu) {
    k_mean_hw<<<dim3(B * Cin), dim3(256), 0, stream>>>(in, inB, Cin, HW, MEAN);
    k_fc<<<g1d((long)B * 4, 64), dim3(64), 0, stream>>>(MEAN, dc.rw, dc.rb, ROUT, B, Cin, 4, 2);
    long N = (long)Cout * Cin;
    k_blend_h<<<g1d((long)B * N, 256), dim3(256), 0, stream>>>(ROUT, dc.w, WBH, B, N);
    dim3 grid((unsigned)((HW + 127) / 128), (unsigned)((Cout + 15) / 16), (unsigned)B);
    k_dc1x1_wmma<<<grid, dim3(128), 0, stream>>>(in, inB, Cin, HW, WBH, out, outB, Cout,
                                                 bn.g, bn.b, bn.m, bn.v, relu);
  };

  // --- dynamic depthwise conv (routing -> blend f32 -> dw conv + BN) ---
  auto dcdw = [&](const float* in, long inB, int Cin, int H, int W,
                  const DCp& dc, const BNp& bn, int K, int stride,
                  float* out, long outB, int Ho, int Wo, int CoutUsed, int relu) {
    int HW = H * W;
    k_mean_hw<<<dim3(B * Cin), dim3(256), 0, stream>>>(in, inB, Cin, HW, MEAN);
    k_fc<<<g1d((long)B * 4, 64), dim3(64), 0, stream>>>(MEAN, dc.rw, dc.rb, ROUT, B, Cin, 4, 2);
    long N = (long)Cin * K * K;
    k_blend_f<<<g1d((long)B * N, 256), dim3(256), 0, stream>>>(ROUT, dc.w, WBF, B, N);
    long total = (long)B * CoutUsed * Ho * Wo;
    k_dwconv<<<g1d(total, 256), dim3(256), 0, stream>>>(
        in, inB, Cin, H, W, nullptr, WBF, K, stride, (K - 1) / 2,
        out, outB, Ho, Wo, CoutUsed, bn.g, bn.b, bn.m, bn.v, relu, total);
  };

  // --- ghost module: primary 1x1 (WMMA) + cheap dyn dw3x3, concat[:oup] ---
  auto ghost = [&](const float* in, long inB, int Cin, int H, int W,
                   const Ghostp& gp, int oup, int relu, float* out) {
    int HW = H * W;
    int i1 = (oup + 1) / 2;
    long outB = (long)oup * HW;
    dc1x1(in, inB, Cin, HW, gp.p, gp.pbn, i1, out, outB, relu);
    int cheap = oup - i1;
    if (cheap > 0)
      dcdw(out, outB, i1, H, W, gp.c, gp.cbn, 3, 1,
           out + (long)i1 * HW, outB, H, W, cheap, relu);
  };

  // ---------------- stem: dynamic 3x3 s2, 3 -> 16, 224 -> 112 ----------------
  {
    k_mean_hw<<<dim3(B * 3), dim3(256), 0, stream>>>(x_in, (long)3 * 224 * 224, 3, 224 * 224, MEAN);
    k_fc<<<g1d((long)B * 4, 64), dim3(64), 0, stream>>>(MEAN, stem.rw, stem.rb, ROUT, B, 3, 4, 2);
    long N = 16L * 3 * 9;
    k_blend_f<<<g1d((long)B * N, 256), dim3(256), 0, stream>>>(ROUT, stem.w, WBF, B, N);
    long total = (long)B * 16 * 112 * 112;
    k_convfull<<<g1d(total, 256), dim3(256), 0, stream>>>(
        x_in, 3, 224, 224, WBF, 3, 2, 1, ACT0, 16, 112, 112,
        stembn.g, stembn.b, stembn.m, stembn.v, 1, total);
  }

  float* curA = ACT0; float* nxt = ACT1;
  int curC = 16, H = 112, W = 112;

  // ---------------- bottleneck blocks ----------------
  for (int i = 0; i < 6; ++i) {
    const BlkCfg& c = BLK[i];
    const Blkp&   p = bp[i];
    int Ho = (c.s == 2) ? H / 2 : H;
    int Wo = (c.s == 2) ? W / 2 : W;

    ghost(curA, (long)curC * H * W, curC, H, W, p.g1, c.mid, 1, MID0);
    float* midc = MID0; int mH = H, mW = W;

    if (c.s > 1) {
      long total = (long)B * c.mid * Ho * Wo;
      k_dwconv<<<g1d(total, 256), dim3(256), 0, stream>>>(
          MID0, (long)c.mid * H * W, c.mid, H, W, p.dw, nullptr, c.k, c.s, (c.k - 1) / 2,
          MID1, (long)c.mid * Ho * Wo, Ho, Wo, c.mid,
          p.dwbn.g, p.dwbn.b, p.dwbn.m, p.dwbn.v, 0, total);
      midc = MID1; mH = Ho; mW = Wo;
    }
    if (c.rd > 0) {
      int mhw = mH * mW;
      k_mean_hw<<<dim3(B * c.mid), dim3(256), 0, stream>>>(midc, (long)c.mid * mhw, c.mid, mhw, MEAN);
      k_fc<<<g1d((long)B * c.rd, 64), dim3(64), 0, stream>>>(MEAN, p.sw1, p.sb1, SE1, B, c.mid, c.rd, 1);
      k_fc<<<g1d((long)B * c.mid, 64), dim3(64), 0, stream>>>(SE1, p.sw2, p.sb2, GATE, B, c.rd, c.mid, 3);
      long total = (long)B * c.mid * mhw;
      k_scale<<<g1d(total, 256), dim3(256), 0, stream>>>(midc, GATE, c.mid, mhw, total);
    }

    ghost(midc, (long)c.mid * mH * mW, c.mid, mH, mW, p.g2, c.out, 0, nxt);

    long outN = (long)B * c.out * Ho * Wo;
    bool idsc = (c.cin == c.out && c.s == 1);
    if (idsc) {
      k_add<<<g1d(outN, 256), dim3(256), 0, stream>>>(nxt, nxt, curA, outN);
    } else {
      dcdw(curA, (long)curC * H * W, curC, H, W, p.s1, p.s1bn, c.k, c.s,
           SH0, (long)curC * Ho * Wo, Ho, Wo, curC, 0);
      dc1x1(SH0, (long)curC * Ho * Wo, curC, Ho * Wo, p.s2, p.s2bn, c.out,
            SH1, (long)c.out * Ho * Wo, 0);
      k_add<<<g1d(outN, 256), dim3(256), 0, stream>>>(nxt, nxt, SH1, outN);
    }
    float* t = curA; curA = nxt; nxt = t;
    curC = c.out; H = Ho; W = Wo;
  }

  // ---------------- final 1x1 (WMMA) + pool + head + classifier ----------------
  dc1x1(curA, (long)curC * H * W, curC, H * W, fin, finbn, 672, MID0, (long)672 * H * W, 1);
  k_mean_hw<<<dim3(B * 672), dim3(256), 0, stream>>>(MID0, (long)672 * H * W, 672, H * W, POOL);
  k_fc<<<g1d((long)B * 1280, 64), dim3(64), 0, stream>>>(POOL, head_w, head_b, FEAT, B, 672, 1280, 1);
  k_fc<<<g1d((long)B * 1000, 64), dim3(64), 0, stream>>>(FEAT, cls_w, cls_b, (float*)d_out, B, 1280, 1000, 0);
}
